// Decoder_87806311400184
// MI455X (gfx1250) — compile-verified, matrix operand order checked by reading, not run on hardware
//
#include <hip/hip_runtime.h>
#include <math.h>

// ---------------------------------------------------------------------------
// Types for CDNA5 WMMA (wave32, 16x16x32 bf16 -> f32)
// ---------------------------------------------------------------------------
typedef __attribute__((ext_vector_type(16))) __bf16 v16bf_t;
typedef __attribute__((ext_vector_type(4)))  __bf16 v4bf_t;
typedef __attribute__((ext_vector_type(8)))  float  v8f_t;

#define H 128
#define NHEAD 8
#define HDIM 16
#define NEG_BIG (-3.0e38f)

// ---------------------------------------------------------------------------
// Kernel 1: context path (gather -> linear_first -> LN -> LSTM step -> LN -> q)
// One workgroup (128 threads) per batch row. Tiny: ~0.3 GFLOP total.
// ---------------------------------------------------------------------------
__device__ __forceinline__ float sigmoidf_(float x) {
    return 1.0f / (1.0f + __expf(-x));
}

__device__ float block_ln128(float y, float g, float bta, float* red) {
    const int i = threadIdx.x;
    red[i] = y; __syncthreads();
    for (int s = 64; s > 0; s >>= 1) { if (i < s) red[i] += red[i + s]; __syncthreads(); }
    const float mu = red[0] * (1.0f / 128.0f);
    __syncthreads();
    const float d = y - mu;
    red[i] = d * d; __syncthreads();
    for (int s = 64; s > 0; s >>= 1) { if (i < s) red[i] += red[i + s]; __syncthreads(); }
    const float var = red[0] * (1.0f / 128.0f);
    __syncthreads();
    return d * rsqrtf(var + 1e-5f) * g + bta;
}

__global__ __launch_bounds__(128) void ctx_q_kernel(
    const float* __restrict__ node_emb, const float* __restrict__ graph_emb,
    const int* __restrict__ first_node, const int* __restrict__ cur_node,
    const float* __restrict__ W_first, const float* __restrict__ b_first,
    const float* __restrict__ g1, const float* __restrict__ beta1,
    const float* __restrict__ g2, const float* __restrict__ beta2,
    const float* __restrict__ W_ih, const float* __restrict__ b_ih,
    const float* __restrict__ b_hh, const float* __restrict__ Wq,
    float* __restrict__ q_out, int B, int N)
{
    const int b = blockIdx.x;
    const int i = threadIdx.x;

    __shared__ float ctx384[384];
    __shared__ float red[128];
    __shared__ float vbuf[128];

    const int fn = first_node[b];
    const int cn = cur_node[b];
    ctx384[i]       = graph_emb[(size_t)b * H + i];
    ctx384[128 + i] = node_emb[((size_t)b * N + fn) * H + i];
    ctx384[256 + i] = node_emb[((size_t)b * N + cn) * H + i];
    __syncthreads();

    // linear_first: out[i] = b + sum_k ctx384[k] * W_first[i,k]   (W_first: [128,384] row-major)
    float y = b_first[i];
    const float* wr = W_first + (size_t)i * 384;
    #pragma unroll 4
    for (int k = 0; k < 384; ++k) y = fmaf(ctx384[k], wr[k], y);

    const float context = block_ln128(y, g1[i], beta1[i], red);
    vbuf[i] = context;              // safe: block_ln128 ends with __syncthreads()
    __syncthreads();

    // LSTM gates (h_prev = 0, c_prev = 0): need i, g, o gates only
    float gi = b_ih[i]       + b_hh[i];
    float gg = b_ih[256 + i] + b_hh[256 + i];
    float go = b_ih[384 + i] + b_hh[384 + i];
    const float* wi_i = W_ih + (size_t)(      i) * H;
    const float* wi_g = W_ih + (size_t)(256 + i) * H;
    const float* wi_o = W_ih + (size_t)(384 + i) * H;
    #pragma unroll 4
    for (int k = 0; k < H; ++k) {
        const float cv = vbuf[k];
        gi = fmaf(cv, wi_i[k], gi);
        gg = fmaf(cv, wi_g[k], gg);
        go = fmaf(cv, wi_o[k], go);
    }
    const float cc = sigmoidf_(gi) * tanhf(gg);
    const float hh = sigmoidf_(go) * tanhf(cc);

    const float ctx2 = block_ln128(hh, g2[i], beta2[i], red);
    vbuf[i] = ctx2;
    __syncthreads();

    // q = ctx2 @ Wq^T
    float q = 0.0f;
    const float* wq = Wq + (size_t)i * H;
    #pragma unroll 4
    for (int k = 0; k < H; ++k) q = fmaf(vbuf[k], wq[k], q);
    q_out[(size_t)b * H + i] = q;
}

// ---------------------------------------------------------------------------
// Stage a 16-node x 128-ch f32 tile from global, converting to bf16 directly
// into WMMA A-fragment layout in LDS:
//   afrag[((ks*32 + lane)*16 + e)] = X[m, 32*ks + kkl]
//   with lane = m + 16*hs, hs = (kkl>>3)&1, e = (kkl&7) + ((kkl>>4)<<3)
// Each thread: 2 coalesced float4 global loads -> 2 packed 8B LDS stores.
// ---------------------------------------------------------------------------
__device__ __forceinline__ void stage_tile_bf16(const float* __restrict__ xb,
                                                int n0, int N, int tid,
                                                __bf16* afrag)
{
    #pragma unroll
    for (int ph = 0; ph < 2; ++ph) {
        const int f    = tid + ph * 256;   // float4 id 0..511
        const int row  = f >> 5;           // node within tile (32 float4 per row)
        const int col4 = f & 31;
        float4 val = make_float4(0.f, 0.f, 0.f, 0.f);
        if (n0 + row < N) val = ((const float4*)(xb + (size_t)(n0 + row) * H))[col4];
        if (n0 + 16 + row < N)  // software prefetch of next tile (global_prefetch_b8)
            __builtin_prefetch((const void*)(xb + (size_t)(n0 + 16 + row) * H + col4 * 4), 0, 0);
        const int kk0 = col4 * 4;          // 4 consecutive kk, same 8-block
        const int ks  = kk0 >> 5;
        const int r   = kk0 & 31;
        const int hsv = (r >> 3) & 1;
        const int e0  = (r & 7) + ((r >> 4) << 3);   // in {0,4,8,12}
        const int dst = (ks * 32 + (row + 16 * hsv)) * 16 + e0;
        v4bf_t pk = { (__bf16)val.x, (__bf16)val.y, (__bf16)val.z, (__bf16)val.w };
        *(v4bf_t*)(&afrag[dst]) = pk;
    }
}

// ---------------------------------------------------------------------------
// Kernel 2: fused K/V projections (WMMA) + flash single-query MHA, then
// pointer logits via the algebraic identity
//   comp[n] = (X_n @ Wlk^T + blk) . ce = X_n . (Wlk^T ce) + blk . ce
// so no second GEMM pass is needed (per-batch X slice stays hot in L2).
// One workgroup (256 threads = 8 wave32) per batch row.
// ---------------------------------------------------------------------------
__global__ __launch_bounds__(256) void decode_kernel(
    const float* __restrict__ node_emb, const int* __restrict__ mask,
    const float* __restrict__ Wk, const float* __restrict__ Wv,
    const float* __restrict__ Wlk, const float* __restrict__ blk,
    const float* __restrict__ Wproj, const float* __restrict__ q_in,
    float* __restrict__ out, int B, int N)
{
    const int b    = blockIdx.x;
    const int tid  = threadIdx.x;
    const int w    = tid >> 5;        // wave 0..7
    const int lane = tid & 31;
    const int c    = lane & 15;       // column within tile
    const int hs   = lane >> 4;       // half-select
    const int j    = 16 * w + c;      // global output channel 0..127
    const int h    = c & 7;           // head (fast dim of channel split: j = e*8 + h)

    __shared__ __align__(16) __bf16 afrag[4 * 32 * 16];  // A-fragments, 4 k-steps (4 KB)
    __shared__ float s_lds[128];      // raw qk partial sums [8 heads][16 nodes] / reduce scratch
    __shared__ float p_lds[128];      // softmax probabilities [8 heads][16 nodes]
    __shared__ float maskf_lds[16];
    __shared__ float scale_lds[NHEAD];
    __shared__ float mrun_lds[NHEAD];
    __shared__ float lrun_lds[NHEAD];
    __shared__ float attout[128];
    __shared__ float ce_lds[128];
    __shared__ float wcomp_lds[128];

    // ---- Preload B-fragments of Wk/Wv for this wave's channel tile (f32 -> bf16)
    v16bf_t bk[4], bv[4];
    #pragma unroll
    for (int ks = 0; ks < 4; ++ks) {
        #pragma unroll
        for (int ee = 0; ee < 16; ++ee) {
            const int kk = 32 * ks + 8 * hs + ee + ((ee >= 8) ? 8 : 0);
            bk[ks][ee] = (__bf16)Wk[(size_t)j * H + kk];
            bv[ks][ee] = (__bf16)Wv[(size_t)j * H + kk];
        }
    }
    const float qj = q_in[(size_t)b * H + j];

    if (tid < NHEAD) { mrun_lds[tid] = NEG_BIG; lrun_lds[tid] = 0.0f; }
    if (tid < 128)   attout[tid] = 0.0f;

    float oh = 0.0f;                   // this lane's 8-node half of oh[h,e]
    const int ntiles = (N + 15) >> 4;
    const float* xb = node_emb + (size_t)b * N * H;
    const v16bf_t* af = (const v16bf_t*)afrag;

    // =============== pass 1: K,V projections + flash attention ===============
    for (int t = 0; t < ntiles; ++t) {
        const int n0 = t * 16;
        __syncthreads();
        stage_tile_bf16(xb, n0, N, tid, afrag);
        if (tid < 16)  maskf_lds[tid] = (n0 + tid < N) ? (float)mask[(size_t)b * N + n0 + tid] : 0.0f;
        if (tid < 128) s_lds[tid] = 0.0f;
        __syncthreads();

        // K and V tiles via WMMA (K-dim 128 = 4 steps of 32)
        v8f_t accK = {}; v8f_t accV = {};
        #pragma unroll
        for (int ks = 0; ks < 4; ++ks) {
            const v16bf_t a = af[ks * 32 + lane];
            accK = __builtin_amdgcn_wmma_f32_16x16x32_bf16(false, a, false, bk[ks], (short)0, accK, false, false);
            accV = __builtin_amdgcn_wmma_f32_16x16x32_bf16(false, a, false, bv[ks], (short)0, accV, false, false);
        }

        // score partials: every lane adds its own q[j]*k[n,j] (no shuffles/branches)
        #pragma unroll
        for (int r = 0; r < 8; ++r)
            atomicAdd(&s_lds[h * 16 + 8 * hs + r], qj * accK[r]);
        __syncthreads();

        // online softmax: ONE thread per head owns the running state.
        // All LDS loads are unconditional (always in-range); tail nodes are
        // squashed with a cndmask select, and exp(NEG_BIG - m) underflows to 0.
        if (tid < NHEAD) {
            const int   hh    = tid;
            const int   valid = N - n0;          // >= 16 except on the last tile
            const float mold  = mrun_lds[hh];
            float sv[16];
            float tmax = NEG_BIG;
            #pragma unroll
            for (int mm = 0; mm < 16; ++mm) {
                float s = fmaf(s_lds[hh * 16 + mm], 0.25f, maskf_lds[mm]);
                s = (mm < valid) ? s : NEG_BIG;
                sv[mm] = s;
                tmax   = fmaxf(tmax, s);
            }
            const float mnew = fmaxf(mold, tmax);
            const float sc   = __expf(mold - mnew);
            float ls = 0.0f;
            #pragma unroll
            for (int mm = 0; mm < 16; ++mm) {
                const float pe = __expf(sv[mm] - mnew);   // underflows to 0 for tail
                p_lds[hh * 16 + mm] = pe;
                ls += pe;
            }
            lrun_lds[hh] = lrun_lds[hh] * sc + ls;
            mrun_lds[hh] = mnew;
            scale_lds[hh] = sc;
        }
        __syncthreads();

        // oh[h,e] half-accumulation: this lane covers its 8 nodes only;
        // the partner half (lane^16) is merged once at the end via atomicAdd.
        const float sc = scale_lds[h];
        float contrib = 0.0f;
        #pragma unroll
        for (int r = 0; r < 8; ++r)
            contrib = fmaf(p_lds[h * 16 + 8 * hs + r], accV[r], contrib);
        oh = oh * sc + contrib;
    }

    // =============== att_out -> context_emb = att_out @ Wproj^T ===============
    __syncthreads();
    {
        const int e = 2 * w + (c >> 3);          // att_out index = h*16 + e (h slow, e fast)
        atomicAdd(&attout[h * 16 + e], oh / lrun_lds[h]);   // both halves merge here
    }
    __syncthreads();
    if (tid < 128) {
        float acc = 0.0f;
        const float* wp = Wproj + (size_t)tid * H;
        #pragma unroll 4
        for (int d = 0; d < H; ++d) acc = fmaf(attout[d], wp[d], acc);
        ce_lds[tid] = acc;
    }
    __syncthreads();

    // =============== wcomp = Wlk^T ce ; bce = blk . ce ===============
    if (tid < 128) {
        float acc = 0.0f;
        #pragma unroll 4
        for (int jj = 0; jj < H; ++jj)
            acc = fmaf(Wlk[(size_t)jj * H + tid], ce_lds[jj], acc);
        wcomp_lds[tid] = acc;
        s_lds[tid] = blk[tid] * ce_lds[tid];
    }
    __syncthreads();
    for (int s = 64; s > 0; s >>= 1) { if (tid < s) s_lds[tid] += s_lds[tid + s]; __syncthreads(); }
    const float bce = s_lds[0];

    // =============== pointer logits: comp[n] = X_n . wcomp + bce ===============
    for (int n = tid; n < N; n += 256) {
        const float4* xr = (const float4*)(xb + (size_t)n * H);
        float acc = bce;
        #pragma unroll 8
        for (int k4 = 0; k4 < 32; ++k4) {
            const float4 xv = xr[k4];
            acc = fmaf(xv.x, wcomp_lds[k4 * 4 + 0], acc);
            acc = fmaf(xv.y, wcomp_lds[k4 * 4 + 1], acc);
            acc = fmaf(xv.z, wcomp_lds[k4 * 4 + 2], acc);
            acc = fmaf(xv.w, wcomp_lds[k4 * 4 + 3], acc);
        }
        const int mk = mask[(size_t)b * N + n];
        float v = 10.0f * tanhf(acc * 0.08838834764831845f);   // 1/sqrt(128)
        out[(size_t)b * N + n] = (mk == 1) ? -__builtin_inff() : v;
    }
}

// ---------------------------------------------------------------------------
// Host launcher
// ---------------------------------------------------------------------------
extern "C" void kernel_launch(void* const* d_in, const int* in_sizes, int n_in,
                              void* d_out, int out_size, void* d_ws, size_t ws_size,
                              hipStream_t stream)
{
    const float* node_emb   = (const float*)d_in[0];
    const float* graph_emb  = (const float*)d_in[1];
    const int*   first_node = (const int*)  d_in[2];
    const int*   cur_node   = (const int*)  d_in[3];
    const int*   mask       = (const int*)  d_in[4];
    const float* W_first    = (const float*)d_in[5];
    const float* b_first    = (const float*)d_in[6];
    const float* g1         = (const float*)d_in[7];
    const float* beta1      = (const float*)d_in[8];
    const float* g2         = (const float*)d_in[9];
    const float* beta2      = (const float*)d_in[10];
    const float* W_ih       = (const float*)d_in[11];
    // d_in[12] = W_hh (unused: h_prev == 0)
    const float* b_ih       = (const float*)d_in[13];
    const float* b_hh       = (const float*)d_in[14];
    const float* Wq         = (const float*)d_in[15];
    const float* Wk         = (const float*)d_in[16];
    const float* Wv         = (const float*)d_in[17];
    const float* Wproj      = (const float*)d_in[18];
    const float* Wlk        = (const float*)d_in[19];
    const float* blk        = (const float*)d_in[20];

    const int B = in_sizes[1] / H;              // graph_embeddings is [B, 128]
    const int N = in_sizes[0] / (B * H);        // node_embeddings is [B, N, 128]

    float* q_ws = (float*)d_ws;                 // [B, 128] query scratch

    ctx_q_kernel<<<B, 128, 0, stream>>>(node_emb, graph_emb, first_node, cur_node,
                                        W_first, b_first, g1, beta1, g2, beta2,
                                        W_ih, b_ih, b_hh, Wq, q_ws, B, N);

    decode_kernel<<<B, 256, 0, stream>>>(node_emb, mask, Wk, Wv, Wlk, blk, Wproj,
                                         q_ws, (float*)d_out, B, N);
}
